// SparseGPAttention_79499844649599
// MI455X (gfx1250) — matrix operand order repaired and checked by reading, not compiled, not run on hardware
//
#include <hip/hip_runtime.h>
#include <stdint.h>
#include <stddef.h>

typedef __bf16 bf16;
typedef __attribute__((ext_vector_type(16))) __bf16 v16bf;
typedef __attribute__((ext_vector_type(8)))  __bf16 bf16x8;
typedef __attribute__((ext_vector_type(8)))  float  v8f;
typedef __attribute__((ext_vector_type(4)))  unsigned int v4u;
typedef __attribute__((ext_vector_type(8)))  int v8i_;
typedef __attribute__((ext_vector_type(4)))  int v4i_;

// ---------------------------------------------------------------------------
// f32 -> bf16 conversion (GEMM operands converted once, reused across tiles)
// ---------------------------------------------------------------------------
__global__ void cvt_f32_bf16(const float* __restrict__ in, bf16* __restrict__ out, int n) {
    int i = blockIdx.x * 256 + threadIdx.x;
    if (i < n) out[i] = (bf16)in[i];
}

// ---------------------------------------------------------------------------
// C[m,n] = sum_k X[m,k] * W[n,k] (+ bias[n])   -- X:[Nr,1024], W:[1024,1024]
// 256 threads / 8 waves; 128x128 C tile; double-buffered 64-wide K chunks.
// A tile: per-lane global_load_async_to_lds_b128 (ASYNCcnt).
// B tile: Tensor Data Mover tensor_load_to_lds, one descriptor per chunk,
//         with HW padding (32 DWORDs data + 4 DWORDs pad = 144B LDS pitch).
// bf16 WMMA, fp32 accumulate.
// ---------------------------------------------------------------------------
__global__ __launch_bounds__(256)
void gemm_bias_wmma(const bf16* __restrict__ X, const bf16* __restrict__ W,
                    const float* __restrict__ bias, float* __restrict__ C)
{
    // padded rows: 72 halves = 144B -> 16-lane column reads hit distinct banks
    __shared__ bf16 tA[2][128 * 72];
    __shared__ bf16 tB[2][128 * 72];

    const int tid  = threadIdx.x;
    const int lane = tid & 31, wid = tid >> 5;
    const int l15  = lane & 15, lg  = lane >> 4;
    const int wm   = wid & 3,   wn  = wid >> 2;     // 4 waves on M, 2 on N
    const int m0   = blockIdx.y * 128;
    const int n0   = blockIdx.x * 128;

    union Frag { v16bf v; bf16x8 h[2]; };

    v8f acc[2][4];
#pragma unroll
    for (int a = 0; a < 2; ++a)
#pragma unroll
        for (int b = 0; b < 4; ++b)
            acc[a][b] = (v8f){0.f,0.f,0.f,0.f,0.f,0.f,0.f,0.f};

    // ---- stage A tile via per-lane async global->LDS copies ----
    auto stageA = [&](int kc, int buf) {
#pragma unroll
        for (int it = 0; it < 4; ++it) {
            int idx = tid + it * 256;                // 1024 chunks of 16B
            int r = idx >> 3, ch = idx & 7;
            const bf16* gp = X + (size_t)(m0 + r) * 1024 + kc + ch * 8;
            uint32_t la = (uint32_t)(size_t)(&tA[buf][r * 72 + ch * 8]);
            asm volatile("global_load_async_to_lds_b128 %0, %1, off"
                         :: "v"(la), "v"((uint64_t)(size_t)gp) : "memory");
        }
    };

    // ---- stage B tile via TDM: 128x64 bf16 tile of W, HW-padded rows ----
    auto stageB = [&](int kc, int buf) {
        if (wid == 0) {
            uint64_t ga = (uint64_t)(size_t)(W + (size_t)n0 * 1024 + kc);
            uint32_t la = (uint32_t)(size_t)(&tB[buf][0]);
            v4u g0;
            g0.x = 1u;                               // count=1, gather off
            g0.y = la;                               // lds_addr
            g0.z = (uint32_t)ga;                     // global_addr[31:0]
            g0.w = (uint32_t)(ga >> 32) | (2u << 30);// global_addr[56:32] | type=2
            v8i_ g1;
            // data_size=2B (1<<16) | pad_enable (1<<20) | pad_interval=32DW
            // (4<<22) | pad_amount=4DW (3<<25)
            g1[0] = (int)((1u << 16) | (1u << 20) | (4u << 22) | (3u << 25));
            g1[1] = (int)(1024u << 16);              // tensor_dim0 = 1024 (bits 79:48)
            g1[2] = (int)(1024u << 16);              // tensor_dim1 = 1024 (bits 111:80)
            g1[3] = (int)(64u << 16);                // tile_dim0 = 64   (bits 127:112)
            g1[4] = 128;                             // tile_dim1 = 128  (bits 143:128)
            g1[5] = 1024;                            // tensor_dim0_stride (bits 207:160)
            g1[6] = 0;
            g1[7] = 0;
            v4i_ gz4 = {0, 0, 0, 0};                 // groups 2/3 unused (2D tensor)
            v8i_ gz8 = {0, 0, 0, 0, 0, 0, 0, 0};     // trailing words (6-arg form)
            __builtin_amdgcn_tensor_load_to_lds(g0, g1, gz4, gz4, gz8, 0);
        }
    };

    auto waitLoads = [&]() {
        asm volatile("s_wait_asynccnt 0x0" ::: "memory");
        __builtin_amdgcn_s_wait_tensorcnt(0);
    };

    auto compute = [&](int buf) {
#pragma unroll
        for (int kk = 0; kk < 64; kk += 32) {
            Frag a[2], bfr[4];
            // A 16x32 frag (ISA 7.12.2): lane group g holds K = e+8g (e<8),
            // K = 8+e+8g (e>=8)  -> two 16B chunks at +8g and +16+8g
#pragma unroll
            for (int mt = 0; mt < 2; ++mt) {
                int r = wm * 32 + mt * 16 + l15;
                a[mt].h[0] = *(const bf16x8*)&tA[buf][r * 72 + kk + 8 * lg];
                a[mt].h[1] = *(const bf16x8*)&tA[buf][r * 72 + kk + 16 + 8 * lg];
            }
            // B 32x16 frag: lane = column n, lane group g holds K = e+8g (e<8), K = 8+e+8g (e>=8)
#pragma unroll
            for (int nt = 0; nt < 4; ++nt) {
                int r = wn * 64 + nt * 16 + l15;
                bfr[nt].h[0] = *(const bf16x8*)&tB[buf][r * 72 + kk + 8 * lg];
                bfr[nt].h[1] = *(const bf16x8*)&tB[buf][r * 72 + kk + 16 + 8 * lg];
            }
#pragma unroll
            for (int mt = 0; mt < 2; ++mt)
#pragma unroll
                for (int nt = 0; nt < 4; ++nt)
                    acc[mt][nt] = __builtin_amdgcn_wmma_f32_16x16x32_bf16(
                        false, a[mt].v, false, bfr[nt].v,
                        (short)0, acc[mt][nt], false, false);
        }
    };

    // ---- software pipeline: prefetch chunk i+1 while computing chunk i ----
    stageA(0, 0);
    stageB(0, 0);
    waitLoads();
    __syncthreads();

    int cur = 0;
    for (int kc = 0; kc < 1024; kc += 64) {
        const bool hasNext = (kc + 64) < 1024;
        if (hasNext) {                               // prefetch into other buffer
            stageA(kc + 64, cur ^ 1);
            stageB(kc + 64, cur ^ 1);
        }
        compute(cur);
        if (hasNext) waitLoads();
        __syncthreads();
        cur ^= 1;
    }

    // epilogue: C layout (f32 16x16): lane 0-15 N=lane; VGPR r -> M = r+8*(lane>=16)
#pragma unroll
    for (int mt = 0; mt < 2; ++mt)
#pragma unroll
        for (int nt = 0; nt < 4; ++nt) {
            int col = n0 + wn * 64 + nt * 16 + l15;
            float bv = bias ? bias[col] : 0.f;
#pragma unroll
            for (int r = 0; r < 8; ++r) {
                int row = m0 + wm * 32 + mt * 16 + r + 8 * lg;
                C[(size_t)row * 1024 + col] = acc[mt][nt][r] + bv;
            }
        }
}

// ---------------------------------------------------------------------------
// zz[i] = ||Z[i,:]||^2
// ---------------------------------------------------------------------------
__global__ void rownorm_kernel(const float* __restrict__ Z, float* __restrict__ zz) {
    __shared__ float red[256];
    int i = blockIdx.x, tid = threadIdx.x;
    float p = 0.f;
    for (int e = tid; e < 1024; e += 256) { float v = Z[(size_t)i * 1024 + e]; p += v * v; }
    red[tid] = p; __syncthreads();
    for (int s = 128; s > 0; s >>= 1) { if (tid < s) red[tid] += red[tid + s]; __syncthreads(); }
    if (tid == 0) zz[i] = red[0];
}

// K_mm = exp(-0.5*(zz_i + zz_j - 2 G)) + jitter*I   (in place on G)
__global__ void rbf_kernel(float* __restrict__ G, const float* __restrict__ zz) {
    int idx = blockIdx.x * 256 + threadIdx.x;
    int i = idx >> 10, j = idx & 1023;
    float v = __expf(-0.5f * (zz[i] + zz[j] - 2.f * G[idx]));
    if (i == j) v += 1e-6f;
    G[idx] = v;
}

// K_nm[b,s] = exp(-0.5*||k[b,s]-Z[s]||^2)   (diagonal kernel, [2,1024])
__global__ void knm_kernel(const float* __restrict__ kp, const float* __restrict__ Z,
                           float* __restrict__ Knm) {
    __shared__ float red[256];
    int row = blockIdx.x, tid = threadIdx.x;        // 0..2047
    int s = row & 1023;
    float p = 0.f;
    for (int e = tid; e < 1024; e += 256) {
        float d = kp[(size_t)row * 1024 + e] - Z[(size_t)s * 1024 + e];
        p += d * d;
    }
    red[tid] = p; __syncthreads();
    for (int st = 128; st > 0; st >>= 1) { if (tid < st) red[tid] += red[tid + st]; __syncthreads(); }
    if (tid == 0) Knm[row] = __expf(-0.5f * red[0]);
}

// ---------------------------------------------------------------------------
// single-workgroup right-looking Cholesky, lower triangle in place
// ---------------------------------------------------------------------------
__global__ __launch_bounds__(1024)
void cholesky_kernel(float* __restrict__ A) {
    __shared__ float diag;
    const int tid = threadIdx.x;
    for (int k = 0; k < 1024; ++k) {
        if (tid == 0) diag = sqrtf(A[(size_t)k * 1024 + k]);
        __syncthreads();
        float d = diag;
        int j = k + 1 + tid;
        if (j < 1024) A[(size_t)j * 1024 + k] /= d;
        if (tid == 0) A[(size_t)k * 1024 + k] = d;
        __syncthreads();
        if (j < 1024) {
            float ljk = A[(size_t)j * 1024 + k];
            for (int i = k + 1; i <= j; ++i)
                A[(size_t)j * 1024 + i] -= ljk * A[(size_t)i * 1024 + k];
        }
        __syncthreads();
    }
}

// cho_solve: K_mm x = K_nm[b]^T  -> Aout[b,:]   (one block per RHS)
__global__ __launch_bounds__(256)
void chosolve_kernel(const float* __restrict__ L, const float* __restrict__ Knm,
                     float* __restrict__ Aout) {
    __shared__ float y[1024];
    __shared__ float red[256];
    const int tid = threadIdx.x;
    const int b = blockIdx.x;
    const float* rhs = Knm + b * 1024;
    for (int i = 0; i < 1024; ++i) {                 // L y = rhs
        float p = 0.f;
        for (int j = tid; j < i; j += 256) p += L[(size_t)i * 1024 + j] * y[j];
        red[tid] = p; __syncthreads();
        for (int s = 128; s > 0; s >>= 1) { if (tid < s) red[tid] += red[tid + s]; __syncthreads(); }
        if (tid == 0) y[i] = (rhs[i] - red[0]) / L[(size_t)i * 1024 + i];
        __syncthreads();
    }
    for (int i = 1023; i >= 0; --i) {                // L^T x = y (in place)
        float p = 0.f;
        for (int j = i + 1 + tid; j < 1024; j += 256) p += L[(size_t)j * 1024 + i] * y[j];
        red[tid] = p; __syncthreads();
        for (int s = 128; s > 0; s >>= 1) { if (tid < s) red[tid] += red[tid + s]; __syncthreads(); }
        if (tid == 0) y[i] = (y[i] - red[0]) / L[(size_t)i * 1024 + i];
        __syncthreads();
    }
    for (int m = tid; m < 1024; m += 256) Aout[b * 1024 + m] = y[m];
}

__global__ void minmax_kernel(const float* __restrict__ Avec, float* __restrict__ mm) {
    __shared__ float rmx[256], rmn[256];
    const int tid = threadIdx.x, b = blockIdx.x;
    float mx = -1e30f, mn = 1e30f;
    for (int m = tid; m < 1024; m += 256) {
        float v = Avec[b * 1024 + m]; mx = fmaxf(mx, v); mn = fminf(mn, v);
    }
    rmx[tid] = mx; rmn[tid] = mn; __syncthreads();
    for (int s = 128; s > 0; s >>= 1) {
        if (tid < s) { rmx[tid] = fmaxf(rmx[tid], rmx[tid + s]); rmn[tid] = fminf(rmn[tid], rmn[tid + s]); }
        __syncthreads();
    }
    if (tid == 0) { mm[b] = rmx[0]; mm[2 + b] = rmn[0]; }
}

// sq_mu[m,d] = sum_h q_mu[h,m,d]
__global__ void sumqmu_kernel(const float* __restrict__ qmu, float* __restrict__ sqmu) {
    int idx = blockIdx.x * 256 + threadIdx.x;        // < 65536
    float s = 0.f;
#pragma unroll
    for (int h = 0; h < 16; ++h) s += qmu[h * 65536 + idx];
    sqmu[idx] = s;
}

// qs[b,hh,s] = dot(qp[b,s,hh*64:...], sq_mu[s,:])
__global__ void qs_kernel(const float* __restrict__ qp, const float* __restrict__ sqmu,
                          float* __restrict__ qs) {
    int idx = blockIdx.x * 256 + threadIdx.x;        // < 32768
    int b = idx >> 14, rem = idx & 16383;
    int hh = rem >> 10, s = rem & 1023;
    const float* qrow = qp + (size_t)(b * 1024 + s) * 1024 + hh * 64;
    const float* mrow = sqmu + s * 64;
    float acc = 0.f;
#pragma unroll
    for (int d = 0; d < 64; ++d) acc += qrow[d] * mrow[d];
    qs[idx] = acc;
}

// ---------------------------------------------------------------------------
// softmax rows: attn[row,:] = softmax(c * A[b,:]); wave-per-row, exp values
// kept in 32 registers, A resident in LDS; also emits rs[row] = sum(attn)
// ---------------------------------------------------------------------------
__global__ __launch_bounds__(256)
void attn_softmax_kernel(const float* __restrict__ qs, const float* __restrict__ Avec,
                         const float* __restrict__ mm, float* __restrict__ attn,
                         float* __restrict__ rs) {
    __shared__ float As[2 * 1024];
    const int tid = threadIdx.x;
    for (int i = tid; i < 2048; i += 256) As[i] = Avec[i];
    __syncthreads();
    const int lane = tid & 31, wid = tid >> 5;
    const int row = blockIdx.x * 8 + wid;            // 0..32767
    const int b = row >> 14;
    const float c = qs[row] * 0.125f;                // / sqrt(64)
    const float rmax = (c >= 0.f) ? c * mm[b] : c * mm[2 + b];
    const float* Ab = &As[b * 1024];
    float ex[32];
    float sum = 0.f;
#pragma unroll
    for (int i = 0; i < 32; ++i) {
        float e = __expf(c * Ab[lane + 32 * i] - rmax);
        ex[i] = e; sum += e;
    }
#pragma unroll
    for (int o = 16; o > 0; o >>= 1) sum += __shfl_xor(sum, o, 32);
    float inv = 1.f / sum;
    float acc = 0.f;
    float* out = attn + (size_t)row * 1024;
#pragma unroll
    for (int i = 0; i < 32; ++i) { float a = ex[i] * inv; out[lane + 32 * i] = a; acc += a; }
#pragma unroll
    for (int o = 16; o > 0; o >>= 1) acc += __shfl_xor(acc, o, 32);
    if (lane == 0) rs[row] = acc;
}

// ov[b,s,e] = (bf16)(v[b,s,e] * rs[b, e/64, s])
__global__ void scalev_kernel(const float* __restrict__ vp, const float* __restrict__ rs,
                              bf16* __restrict__ ov) {
    int idx = blockIdx.x * 256 + threadIdx.x;        // < 2097152
    int b = idx >> 20, rem = idx & 1048575;
    int s = rem >> 10, e = rem & 1023;
    int hh = e >> 6;
    ov[idx] = (bf16)(vp[idx] * rs[((b * 16 + hh) << 10) + s]);
}

// ---------------------------------------------------------------------------
extern "C" void kernel_launch(void* const* d_in, const int* in_sizes, int n_in,
                              void* d_out, int out_size, void* d_ws, size_t ws_size,
                              hipStream_t stream)
{
    (void)in_sizes; (void)n_in; (void)out_size; (void)ws_size;
    const float* query = (const float*)d_in[0];
    const float* key_  = (const float*)d_in[1];
    const float* value = (const float*)d_in[2];
    const float* Z     = (const float*)d_in[3];
    const float* qmu   = (const float*)d_in[4];
    const float* Wq = (const float*)d_in[5];  const float* bq = (const float*)d_in[6];
    const float* Wk = (const float*)d_in[7];  const float* bk = (const float*)d_in[8];
    const float* Wv = (const float*)d_in[9];  const float* bv = (const float*)d_in[10];
    const float* Wo = (const float*)d_in[11]; const float* bo = (const float*)d_in[12];

    float* out  = (float*)d_out;                     // [2,1024,1024]
    float* attn = out + (size_t)2 * 1024 * 1024;     // [2,16,1024,1024]

    char* w = (char*)d_ws; size_t off = 0;
    auto take = [&](size_t bytes) -> void* {
        void* p = w + off;
        off += bytes; off = (off + 255) & ~(size_t)255;
        return p;
    };
    bf16*  XB   = (bf16*) take((size_t)2048 * 1024 * 2);   // X operand (reused)
    bf16*  WB   = (bf16*) take((size_t)1024 * 1024 * 2);   // W operand (reused)
    float* QP   = (float*)take((size_t)2048 * 1024 * 4);
    float* KP   = (float*)take((size_t)2048 * 1024 * 4);
    float* VP   = (float*)take((size_t)2048 * 1024 * 4);
    float* G    = (float*)take((size_t)1024 * 1024 * 4);   // Gram -> K_mm -> L
    float* zz   = (float*)take(1024 * 4);
    float* Knm  = (float*)take(2048 * 4);
    float* Avec = (float*)take(2048 * 4);
    float* mmb  = (float*)take(4 * 4);
    float* sqmu = (float*)take(65536 * 4);
    float* qsb  = (float*)take(32768 * 4);
    float* rsb  = (float*)take(32768 * 4);

    // q/k/v projections (bf16 WMMA, fp32 accumulate)
    cvt_f32_bf16<<<8192, 256, 0, stream>>>(query, XB, 2048 * 1024);
    cvt_f32_bf16<<<4096, 256, 0, stream>>>(Wq, WB, 1024 * 1024);
    gemm_bias_wmma<<<dim3(8, 16), 256, 0, stream>>>(XB, WB, bq, QP);

    cvt_f32_bf16<<<8192, 256, 0, stream>>>(key_, XB, 2048 * 1024);
    cvt_f32_bf16<<<4096, 256, 0, stream>>>(Wk, WB, 1024 * 1024);
    gemm_bias_wmma<<<dim3(8, 16), 256, 0, stream>>>(XB, WB, bk, KP);

    cvt_f32_bf16<<<8192, 256, 0, stream>>>(value, XB, 2048 * 1024);
    cvt_f32_bf16<<<4096, 256, 0, stream>>>(Wv, WB, 1024 * 1024);
    gemm_bias_wmma<<<dim3(8, 16), 256, 0, stream>>>(XB, WB, bv, VP);

    // K_mm = rbf(Z,Z) + jitter*I  (Gram via WMMA, then transform)
    cvt_f32_bf16<<<4096, 256, 0, stream>>>(Z, XB, 1024 * 1024);
    gemm_bias_wmma<<<dim3(8, 8), 256, 0, stream>>>(XB, XB, nullptr, G);
    rownorm_kernel<<<1024, 256, 0, stream>>>(Z, zz);
    rbf_kernel<<<4096, 256, 0, stream>>>(G, zz);

    // diagonal K_nm, Cholesky, solves, softmax row stats
    knm_kernel<<<2048, 256, 0, stream>>>(KP, Z, Knm);
    cholesky_kernel<<<1, 1024, 0, stream>>>(G);
    chosolve_kernel<<<2, 256, 0, stream>>>(G, Knm, Avec);
    minmax_kernel<<<2, 256, 0, stream>>>(Avec, mmb);

    // rank-separable attention
    sumqmu_kernel<<<256, 256, 0, stream>>>(qmu, sqmu);
    qs_kernel<<<128, 256, 0, stream>>>(QP, sqmu, qsb);
    attn_softmax_kernel<<<4096, 256, 0, stream>>>(qsb, Avec, mmb, attn, rsb);

    // out = (v * sum(attn)) @ Wo^T + bo
    scalev_kernel<<<8192, 256, 0, stream>>>(VP, rsb, XB);
    cvt_f32_bf16<<<4096, 256, 0, stream>>>(Wo, WB, 1024 * 1024);
    gemm_bias_wmma<<<dim3(8, 16), 256, 0, stream>>>(XB, WB, bo, out);
}
